// GINConv_1597727834589
// MI455X (gfx1250) — matrix operand order repaired and checked by reading, not compiled, not run on hardware
//
#include <hip/hip_runtime.h>
#include <hip/hip_bf16.h>

typedef __attribute__((ext_vector_type(16))) __bf16 v16bf;
typedef __attribute__((ext_vector_type(8)))  __bf16 v8bf;
typedef __attribute__((ext_vector_type(8)))  float  v8f;
typedef __attribute__((ext_vector_type(4)))  unsigned int u32x4;
typedef __attribute__((ext_vector_type(8)))  int i32x8;
typedef __attribute__((ext_vector_type(4)))  int i32x4;

#define NNODES 50000
#define NEDGES 800000
#define TM 64           // rows per tile
#define LDX 264         // padded bf16 LDS stride (256 + 8) -> breaks bank conflicts
// sX + sT + sW1 + sW2 (bf16) + sB2(128) + sSS(512) floats
#define SMEM_BYTES (((TM + TM + 256 + 128) * LDX * 2) + (640 * 4))   // 272896 B < 320KB/WGP

// ---------------------------------------------------------------------------
// TDM: DMA a [rows][256] bf16 weight matrix from global into LDS, inserting
// 16B of padding after every 512B row (-> LDS stride = 264 bf16 = LDX).
// D# layout per CDNA5 ISA 8.3/8.4: group0 {count=1, lds_addr, global_addr,
// type=2}; group1 {data_size=2B, pad_en, pad_interval=128dw, pad_amount=4dw,
// tensor_dim0=256, tensor_dim1=rows, tile=256 x rows, dim0_stride=256}.
// ---------------------------------------------------------------------------
__device__ __forceinline__ void tdm_load_weights(const __bf16* gsrc, const __bf16* ldst,
                                                 int rows) {
    unsigned long long ga = (unsigned long long)(uintptr_t)gsrc;
    unsigned int lds = (unsigned int)(uintptr_t)ldst;   // low 32b of generic = LDS byte offset
    u32x4 g0;
    g0[0] = 1u;                                          // count=1 (valid), no gather
    g0[1] = lds;                                         // lds_addr [63:32]
    g0[2] = (unsigned int)(ga & 0xFFFFFFFFu);            // global_addr [95:64]
    g0[3] = (unsigned int)((ga >> 32) & 0x01FFFFFFu)     // global_addr [120:96]
          | 0x80000000u;                                 // type=2 ("image") [127:126]
    i32x8 g1;
    g1[0] = (int)((1u << 16)     // data_size = 2 bytes
                | (1u << 20)     // pad_enable
                | (6u << 22)     // pad_interval: 2^(6+1)=128 DWORDs (=512B row)
                | (3u << 25));   // pad_amount: 4 DWORDs (=16B)
    g1[1] = (int)(256u << 16);                           // tensor_dim0 = 256 (bits 79:48)
    g1[2] = (int)(((unsigned)rows & 0xFFFFu) << 16);     // tensor_dim1 = rows (bits 111:80)
    g1[3] = (int)(256u << 16);                           // tile_dim0 = 256 (bits 127:112)
    g1[4] = (int)((unsigned)rows & 0xFFFFu);             // tile_dim1 = rows; tile_dim2 = 0
    g1[5] = 256;                                         // tensor_dim0_stride = 256
    g1[6] = 0;
    g1[7] = 0;
    i32x4 z4 = {0, 0, 0, 0};
#if __clang_major__ >= 23
    i32x8 z8 = {0, 0, 0, 0, 0, 0, 0, 0};
    __builtin_amdgcn_tensor_load_to_lds(g0, g1, z4, z4, z8, 0);
#else
    __builtin_amdgcn_tensor_load_to_lds(g0, g1, z4, z4, 0);
#endif
}

// ---------------------------------------------------------------------------
// Fused 2-layer MLP GEMM.  MODE 0: layer1 only, emit per-tile column sum/sumsq
// partials (BatchNorm batch stats).  MODE 1: layer1 -> scale/shift -> ReLU ->
// layer2 (+bias) -> out.  GATHER 1: X = [h[src]+h[dst], e] built on the fly.
// GATHER 0: X = [sum_h[row], sum_e[row]].
// Wave split: each of 8 waves owns a 32-col strip across ALL 64 rows ->
// unique B-fragments per wave, full columns in-register for BN stats.
// ---------------------------------------------------------------------------
template <int MODE, int GATHER>
__global__ void __launch_bounds__(256)
mlp_gemm(const float* __restrict__ pA,      // gather: h      direct: sum_h
         const float* __restrict__ pB,      // gather: e      direct: sum_e
         const int*   __restrict__ src,
         const int*   __restrict__ dst,
         int M,
         const __bf16* __restrict__ W1T,    // [256][256] bf16 (N-major)
         const __bf16* __restrict__ W2T,    // [128][256] bf16 (N-major)
         const float* __restrict__ SS,      // [512] scale(256),shift(256)  (MODE 1)
         const float* __restrict__ bias2,   // [128]                        (MODE 1)
         float* __restrict__ out,           // [M][128]                     (MODE 1)
         float* __restrict__ statsP)        // [tiles][512]                 (MODE 0)
{
    extern __shared__ __align__(16) char smem_raw[];
    __bf16* sX  = (__bf16*)smem_raw;       // [64][264]
    __bf16* sT  = sX  + TM  * LDX;         // [64][264]   layer-2 input
    __bf16* sW1 = sT  + TM  * LDX;         // [256][264]  W1^T
    __bf16* sW2 = sW1 + 256 * LDX;         // [128][264]  W2^T
    float*  sF  = (float*)(sW2 + 128 * LDX);
    float*  sB2 = sF;                      // [128]
    float*  sSS = sF + 128;                // [512]

    const int tid  = threadIdx.x;
    const int lane = tid & 31;
    const int wave = tid >> 5;             // 0..7 : owns cols [wave*32, wave*32+32)
    const int lcol = lane & 15;
    const int lhi  = lane >> 4;
    const int nTiles = (M + TM - 1) / TM;
    const int gstep  = gridDim.x * TM;

    // ---- stage weights via Tensor Data Mover (wave 0 issues, all wait) ----
    if (tid < 32) {
        tdm_load_weights(W1T, sW1, 256);
        tdm_load_weights(W2T, sW2, 128);
        __builtin_amdgcn_s_wait_tensorcnt(0);
    }
    if (MODE == 1) {
        if (tid < 128) sB2[tid] = bias2[tid];
        for (int i = tid; i < 512; i += 256) sSS[i] = SS[i];
    }
    __syncthreads();

    for (int tile = blockIdx.x; tile < nTiles; tile += gridDim.x) {
        const int r0 = tile * TM;

        // ---- stage X tile [64][256] fp32 -> bf16 LDS (+ prefetch next tile) ----
        for (int it = tid; it < TM * 32; it += 256) {
            int row = it >> 5;
            int c0  = (it & 31) << 3;      // 8 cols per chunk
            int gr  = r0 + row;
            int pg  = gr + gstep;          // next tile this block will touch
            float v[8];
            if (gr < M) {
                if (GATHER) {
                    if (c0 < 128) {
                        int s = src[gr], d = dst[gr];
                        const float4* hs = (const float4*)&pA[(size_t)s * 128 + c0];
                        const float4* hd = (const float4*)&pA[(size_t)d * 128 + c0];
                        float4 a0 = hs[0], a1 = hs[1], b0 = hd[0], b1 = hd[1];
                        v[0]=a0.x+b0.x; v[1]=a0.y+b0.y; v[2]=a0.z+b0.z; v[3]=a0.w+b0.w;
                        v[4]=a1.x+b1.x; v[5]=a1.y+b1.y; v[6]=a1.z+b1.z; v[7]=a1.w+b1.w;
                        if (pg < M) { __builtin_prefetch(&src[pg], 0, 2);
                                      __builtin_prefetch(&dst[pg], 0, 2); }
                    } else {
                        const float4* pe = (const float4*)&pB[(size_t)gr * 128 + (c0 - 128)];
                        float4 a0 = pe[0], a1 = pe[1];
                        v[0]=a0.x; v[1]=a0.y; v[2]=a0.z; v[3]=a0.w;
                        v[4]=a1.x; v[5]=a1.y; v[6]=a1.z; v[7]=a1.w;
                        if (pg < M) __builtin_prefetch(&pB[(size_t)pg * 128 + (c0 - 128)], 0, 2);
                    }
                } else {
                    const float4* p = (c0 < 128)
                        ? (const float4*)&pA[(size_t)gr * 128 + c0]
                        : (const float4*)&pB[(size_t)gr * 128 + (c0 - 128)];
                    float4 a0 = p[0], a1 = p[1];
                    v[0]=a0.x; v[1]=a0.y; v[2]=a0.z; v[3]=a0.w;
                    v[4]=a1.x; v[5]=a1.y; v[6]=a1.z; v[7]=a1.w;
                    if (pg < M) __builtin_prefetch((c0 < 128)
                        ? &pA[(size_t)pg * 128 + c0]
                        : &pB[(size_t)pg * 128 + (c0 - 128)], 0, 2);
                }
            } else {
                #pragma unroll
                for (int i = 0; i < 8; i++) v[i] = 0.0f;
            }
            v8bf o;
            #pragma unroll
            for (int i = 0; i < 8; i++) o[i] = (__bf16)v[i];
            *(v8bf*)&sX[row * LDX + c0] = o;
        }
        __syncthreads();

        // ---- layer 1: [64,256] x [256,256]; wave computes cols wave*32..+32,
        //      all 4 row-blocks; acc[mb*2+nt] ----
        v8f acc[8] = {};
        for (int k0 = 0; k0 < 256; k0 += 32) {
            v16bf af[4];
            #pragma unroll
            for (int m = 0; m < 4; m++) {
                int arow = m * 16 + lcol;
                v8bf alo = *(const v8bf*)&sX[arow * LDX + k0 + lhi * 8];
                v8bf ahi = *(const v8bf*)&sX[arow * LDX + k0 + 16 + lhi * 8];
                af[m] = __builtin_shufflevector(alo, ahi,
                    0,1,2,3,4,5,6,7,8,9,10,11,12,13,14,15);
            }
            #pragma unroll
            for (int nt = 0; nt < 2; nt++) {
                int col = wave * 32 + nt * 16 + lcol;
                v8bf blo = *(const v8bf*)&sW1[col * LDX + k0 + lhi * 8];
                v8bf bhi = *(const v8bf*)&sW1[col * LDX + k0 + 16 + lhi * 8];
                v16bf bf = __builtin_shufflevector(blo, bhi,
                    0,1,2,3,4,5,6,7,8,9,10,11,12,13,14,15);
                #pragma unroll
                for (int m = 0; m < 4; m++)
                    acc[m * 2 + nt] = __builtin_amdgcn_wmma_f32_16x16x32_bf16(
                        false, af[m], false, bf, (short)0, acc[m * 2 + nt], false, false);
            }
        }

        if (MODE == 0) {
            // ---- BN batch stats: full column lives in this wave's registers ----
            #pragma unroll
            for (int nt = 0; nt < 2; nt++) {
                float s = 0.f, q = 0.f;
                #pragma unroll
                for (int m = 0; m < 4; m++)
                    #pragma unroll
                    for (int r = 0; r < 8; r++) {
                        float y = acc[m * 2 + nt][r]; s += y; q += y * y;
                    }
                s += __shfl_xor(s, 16, 32);
                q += __shfl_xor(q, 16, 32);
                if (lhi == 0) {
                    int col = wave * 32 + nt * 16 + lcol;
                    statsP[(size_t)tile * 512 + col]       = s;
                    statsP[(size_t)tile * 512 + 256 + col] = q;
                }
            }
            __syncthreads();   // before next tile overwrites sX
        } else {
            // ---- normalize (folded scale/shift), ReLU, bf16 -> sT ----
            #pragma unroll
            for (int nt = 0; nt < 2; nt++) {
                int col = wave * 32 + nt * 16 + lcol;
                float sc = sSS[col], sh = sSS[256 + col];
                #pragma unroll
                for (int m = 0; m < 4; m++)
                    #pragma unroll
                    for (int r = 0; r < 8; r++) {
                        float y = fmaxf(acc[m * 2 + nt][r] * sc + sh, 0.0f);
                        int row = m * 16 + r + lhi * 8;
                        sT[row * LDX + col] = (__bf16)y;
                    }
            }
            __syncthreads();

            // ---- layer 2: [64,256] x [256,128]; wave owns cols wave*16..+16 ----
            v8f acc2[4] = {};
            const int col2 = wave * 16 + lcol;
            for (int k0 = 0; k0 < 256; k0 += 32) {
                v8bf blo = *(const v8bf*)&sW2[col2 * LDX + k0 + lhi * 8];
                v8bf bhi = *(const v8bf*)&sW2[col2 * LDX + k0 + 16 + lhi * 8];
                v16bf bf = __builtin_shufflevector(blo, bhi,
                    0,1,2,3,4,5,6,7,8,9,10,11,12,13,14,15);
                #pragma unroll
                for (int m = 0; m < 4; m++) {
                    int arow = m * 16 + lcol;
                    v8bf alo = *(const v8bf*)&sT[arow * LDX + k0 + lhi * 8];
                    v8bf ahi = *(const v8bf*)&sT[arow * LDX + k0 + 16 + lhi * 8];
                    v16bf af = __builtin_shufflevector(alo, ahi,
                        0,1,2,3,4,5,6,7,8,9,10,11,12,13,14,15);
                    acc2[m] = __builtin_amdgcn_wmma_f32_16x16x32_bf16(
                        false, af, false, bf, (short)0, acc2[m], false, false);
                }
            }
            {
                float b = sB2[col2];
                #pragma unroll
                for (int m = 0; m < 4; m++)
                    #pragma unroll
                    for (int r = 0; r < 8; r++) {
                        int gr = r0 + m * 16 + r + lhi * 8;
                        if (gr < M) out[(size_t)gr * 128 + col2] = acc2[m][r] + b;
                    }
            }
            __syncthreads();
        }
    }
}

// ---------------------------------------------------------------------------
__global__ void zero_f32(float* p, int n) {
    int i = blockIdx.x * blockDim.x + threadIdx.x;
    if (i < n) p[i] = 0.0f;
}

// W row-major [K][N] fp32 -> WT [N][K] bf16
__global__ void pack_wT(const float* __restrict__ W, __bf16* __restrict__ WT, int K, int N) {
    int i = blockIdx.x * blockDim.x + threadIdx.x;
    if (i < K * N) { int n = i / K, k = i % K; WT[i] = (__bf16)W[k * N + n]; }
}

// wave-per-edge: dstbuf[out_rows[e]][c] += vals[(in_rows?in_rows[e]:e)][c]
__global__ void scatter_add(const float* __restrict__ vals, const int* __restrict__ in_rows,
                            const int* __restrict__ out_rows, float* dstbuf, int E) {
    int idx  = blockIdx.x * blockDim.x + threadIdx.x;
    int e    = idx >> 5;
    int lane = idx & 31;
    if (e >= E) return;
    int vr = in_rows ? in_rows[e] : e;
    int orow = out_rows[e];
    float4 v = *(const float4*)&vals[(size_t)vr * 128 + lane * 4];
    float* d = &dstbuf[(size_t)orow * 128 + lane * 4];
    atomicAdd(d + 0, v.x); atomicAdd(d + 1, v.y);
    atomicAdd(d + 2, v.z); atomicAdd(d + 3, v.w);
}

__global__ void reduce_stats1(const float* __restrict__ P, int nTiles, float* __restrict__ P2) {
    int b = blockIdx.x;              // 0..49
    int t = threadIdx.x;             // 0..511
    int chunk = (nTiles + 49) / 50;
    int lo = b * chunk;
    int hi = lo + chunk; if (hi > nTiles) hi = nTiles;
    float s = 0.f;
    for (int i = lo; i < hi; i++) s += P[(size_t)i * 512 + t];
    P2[b * 512 + t] = s;
}

__global__ void reduce_stats2(const float* __restrict__ P2, const float* __restrict__ g,
                              const float* __restrict__ beta, float invCnt,
                              float* __restrict__ SS) {
    int t = threadIdx.x;             // 0..255
    float s = 0.f, q = 0.f;
    for (int b = 0; b < 50; b++) { s += P2[b * 512 + t]; q += P2[b * 512 + 256 + t]; }
    float mu   = s * invCnt;
    float var  = fmaxf(q * invCnt - mu * mu, 0.0f);
    float rstd = rsqrtf(var + 1e-5f);
    float sc   = g[t] * rstd;
    SS[t]       = sc;
    SS[256 + t] = beta[t] - mu * sc;
}

// ---------------------------------------------------------------------------
extern "C" void kernel_launch(void* const* d_in, const int* in_sizes, int n_in,
                              void* d_out, int out_size, void* d_ws, size_t ws_size,
                              hipStream_t stream) {
    const float* h      = (const float*)d_in[0];
    const float* e      = (const float*)d_in[1];
    const int*   src    = (const int*)  d_in[2];
    const int*   dst    = (const int*)  d_in[3];
    const float* W1b    = (const float*)d_in[4];
    // d_in[5] = b1b : cancels exactly under training-mode BatchNorm
    const float* g1b    = (const float*)d_in[6];
    const float* beta1b = (const float*)d_in[7];
    const float* W2b    = (const float*)d_in[8];
    const float* b2b    = (const float*)d_in[9];
    const float* W1a    = (const float*)d_in[10];
    const float* g1a    = (const float*)d_in[12];
    const float* beta1a = (const float*)d_in[13];
    const float* W2a    = (const float*)d_in[14];
    const float* b2a    = (const float*)d_in[15];

    char* ws = (char*)d_ws;
    __bf16* W1bT   = (__bf16*)(ws + 0);          // 131072 B
    __bf16* W2bT   = (__bf16*)(ws + 131072);     //  65536 B
    __bf16* W1aT   = (__bf16*)(ws + 196608);     // 131072 B
    __bf16* W2aT   = (__bf16*)(ws + 327680);     //  65536 B
    float*  sum_h  = (float*)(ws + 393216);      // 25.6 MB
    float*  sum_e  = (float*)(ws + 25993216);    // 25.6 MB
    float*  statsP = (float*)(ws + 51593216);    // 25.6 MB (12500 tiles * 512)
    float*  P2     = (float*)(ws + 77193216);    // 102400 B
    float*  SSb    = (float*)(ws + 77295616);    // 2048 B
    float*  SSa    = (float*)(ws + 77297664);    // 2048 B

    float* h_new = (float*)d_out;
    float* e_new = (float*)d_out + (size_t)NNODES * 128;

    (void)hipFuncSetAttribute((const void*)&mlp_gemm<0,1>,
        hipFuncAttributeMaxDynamicSharedMemorySize, SMEM_BYTES);
    (void)hipFuncSetAttribute((const void*)&mlp_gemm<1,1>,
        hipFuncAttributeMaxDynamicSharedMemorySize, SMEM_BYTES);
    (void)hipFuncSetAttribute((const void*)&mlp_gemm<0,0>,
        hipFuncAttributeMaxDynamicSharedMemorySize, SMEM_BYTES);
    (void)hipFuncSetAttribute((const void*)&mlp_gemm<1,0>,
        hipFuncAttributeMaxDynamicSharedMemorySize, SMEM_BYTES);

    // 0) zero scatter targets (sum_h and sum_e are contiguous)
    zero_f32<<<(2 * NNODES * 128 + 255) / 256, 256, 0, stream>>>(sum_h, 2 * NNODES * 128);

    // 1) pack weights to bf16, N-major
    pack_wT<<<(256 * 256 + 255) / 256, 256, 0, stream>>>(W1b, W1bT, 256, 256);
    pack_wT<<<(256 * 128 + 255) / 256, 256, 0, stream>>>(W2b, W2bT, 256, 128);
    pack_wT<<<(256 * 256 + 255) / 256, 256, 0, stream>>>(W1a, W1aT, 256, 256);
    pack_wT<<<(256 * 128 + 255) / 256, 256, 0, stream>>>(W2a, W2aT, 256, 128);

    // 2) sum_h[dst] += h[src]
    scatter_add<<<(NEDGES * 32 + 255) / 256, 256, 0, stream>>>(h, src, dst, sum_h, NEDGES);

    const int grid = 1024;
    // 3) bond MLP: stats pass -> fold BN -> full pass (writes e_new)
    mlp_gemm<0,1><<<grid, 256, SMEM_BYTES, stream>>>(h, e, src, dst, NEDGES,
        W1bT, W2bT, nullptr, nullptr, nullptr, statsP);
    reduce_stats1<<<50, 512, 0, stream>>>(statsP, (NEDGES + TM - 1) / TM, P2);
    reduce_stats2<<<1, 256, 0, stream>>>(P2, g1b, beta1b, 1.0f / NEDGES, SSb);
    mlp_gemm<1,1><<<grid, 256, SMEM_BYTES, stream>>>(h, e, src, dst, NEDGES,
        W1bT, W2bT, SSb, b2b, e_new, nullptr);

    // 4) sum_e[dst] += e_new
    scatter_add<<<(NEDGES * 32 + 255) / 256, 256, 0, stream>>>(e_new, nullptr, dst, sum_e, NEDGES);

    // 5) atom MLP: stats pass -> fold BN -> full pass (writes h_new)
    mlp_gemm<0,0><<<grid, 256, SMEM_BYTES, stream>>>(sum_h, sum_e, nullptr, nullptr, NNODES,
        W1aT, W2aT, nullptr, nullptr, nullptr, statsP);
    reduce_stats1<<<50, 512, 0, stream>>>(statsP, (NNODES + TM - 1) / TM, P2);
    reduce_stats2<<<1, 256, 0, stream>>>(P2, g1a, beta1a, 1.0f / NNODES, SSa);
    mlp_gemm<1,0><<<grid, 256, SMEM_BYTES, stream>>>(sum_h, sum_e, nullptr, nullptr, NNODES,
        W1aT, W2aT, SSa, b2a, h_new, nullptr);
}
